// BackgroundFirstSourceFieldEEG_34359738595
// MI455X (gfx1250) — matrix operand order repaired and checked by reading, not compiled, not run on hardware
//
#include <hip/hip_runtime.h>
#include <hip/hip_bf16.h>
#include <stddef.h>

// ---------------------------------------------------------------------------
// Problem constants (from reference)
// ---------------------------------------------------------------------------
#define BB   4
#define NN   32
#define QN   512
#define KN   128
#define DM   512
#define NH   8
#define HD   64
#define ATT_SCALE 0.125f               // 64^-0.5
#define DIST_COEF (-200.0f)            // -1/(2*0.05^2)
#define LEAD_COEF 0.35355339059327373f // 1/sqrt(8)

// ---------------------------------------------------------------------------
// bf16 helpers / WMMA vector types (gfx1250 wave32 WMMA)
// ---------------------------------------------------------------------------
typedef __attribute__((ext_vector_type(16))) __bf16 bf16x16;
typedef __attribute__((ext_vector_type(8)))  __bf16 bf16x8;
typedef __attribute__((ext_vector_type(8)))  float  v8f;
typedef __attribute__((ext_vector_type(4)))  int    i32x4;

__device__ __forceinline__ __bf16 f2bf(float f) {
    union { float f; unsigned u; } v; v.f = f;
    unsigned r = v.u + 0x7FFFu + ((v.u >> 16) & 1u);   // round-to-nearest-even
    unsigned short h = (unsigned short)(r >> 16);
    return __builtin_bit_cast(__bf16, h);
}

// ---------------------------------------------------------------------------
// Async global->LDS staging (GLOBAL_LOAD_ASYNC_TO_LDS_B128, ASYNCcnt-tracked).
// Builtin signature (from probe diagnostic): (v4i32 global*, v4i32 lds*, imm, imm)
// Guarded so the file compiles (VGPR fallback) if the builtins are absent.
// ---------------------------------------------------------------------------
#if defined(__has_builtin)
#if __has_builtin(__builtin_amdgcn_global_load_async_to_lds_b128)
#define HAVE_ASYNC_LDS 1
#endif
#endif

__device__ __forceinline__ void async_copy_b128(void* lds_dst, const void* gsrc) {
#if defined(HAVE_ASYNC_LDS)
    __builtin_amdgcn_global_load_async_to_lds_b128(
        (__attribute__((address_space(1))) i32x4*)gsrc,
        (__attribute__((address_space(3))) i32x4*)lds_dst, 0, 0);
#else
    *(bf16x8*)lds_dst = *(const bf16x8*)gsrc;
#endif
}

__device__ __forceinline__ void wait_async_lds() {
#if defined(HAVE_ASYNC_LDS)
#if __has_builtin(__builtin_amdgcn_s_wait_asynccnt)
    __builtin_amdgcn_s_wait_asynccnt(0);
#else
    asm volatile("s_wait_asynccnt 0x0" ::: "memory");
#endif
#endif
}

// A fragment: 16(M) x 32(K) bf16, LDS row-major with leading dim `ld`.
// ISA layout: lanes 0-15 M=lane; VGPR0-3 hold K = hi*8 + 0..7, VGPR4-7 hold
// K = 16 + hi*8 + 0..7 where hi = lane>>4.
__device__ __forceinline__ bf16x16 frag_a(const __bf16* base, int ld, int row0, int k0) {
    const int lane = threadIdx.x & 31;
    const int l = lane & 15, hi = lane >> 4;
    const __bf16* p = base + (row0 + l) * ld + k0;
    bf16x8 lo = *(const bf16x8*)(p + hi * 8);
    bf16x8 hp = *(const bf16x8*)(p + 16 + hi * 8);
    bf16x16 a;
#pragma unroll
    for (int i = 0; i < 8; ++i) { a[i] = lo[i]; a[8 + i] = hp[i]; }
    return a;
}

// B fragment: 32(K) x 16(N) bf16, where LDS stores B transposed: base[n][k]
// (leading dim `ld` over k). Same K layout as A: VGPR0-3 hold K = hi*8 + 0..7,
// VGPR4-7 hold K = 16 + hi*8 + 0..7.
__device__ __forceinline__ bf16x16 frag_bt(const __bf16* base, int ld, int col0, int k0) {
    const int lane = threadIdx.x & 31;
    const int l = lane & 15, hi = lane >> 4;
    const __bf16* p = base + (col0 + l) * ld + k0 + hi * 8;
    bf16x8 lo = *(const bf16x8*)(p);
    bf16x8 hp = *(const bf16x8*)(p + 16);
    bf16x16 b;
#pragma unroll
    for (int i = 0; i < 8; ++i) { b[i] = lo[i]; b[8 + i] = hp[i]; }
    return b;
}

__device__ __forceinline__ v8f wmma_bf16(bf16x16 a, bf16x16 b, v8f c) {
    return __builtin_amdgcn_wmma_f32_16x16x32_bf16(
        /*neg_a=*/false, a, /*neg_b=*/false, b,
        /*c_mod=*/(short)0, c, /*reuse_a=*/false, /*reuse_b=*/false);
}

// ---------------------------------------------------------------------------
// Kernel 1: positional bias  bias[b,q,k] = -200*dist2 + (qb . kb)/sqrt(8)
// ---------------------------------------------------------------------------
__global__ __launch_bounds__(256) void bias_kernel(
    const float* __restrict__ qpos, const float* __restrict__ kpos,
    const float* __restrict__ Wqb,  const float* __restrict__ Wkb,
    float* __restrict__ biasB)
{
    const int idx = blockIdx.x * 256 + threadIdx.x;   // < B*Q*K = 262144
    const int k = idx & (KN - 1);
    const int q = (idx >> 7) & (QN - 1);
    const int b = idx >> 16;
    const float* qp = qpos + ((size_t)b * QN + q) * 3;
    const float* kp = kpos + ((size_t)b * KN + k) * 3;
    const float q0 = qp[0], q1 = qp[1], q2 = qp[2];
    const float k0 = kp[0], k1 = kp[1], k2 = kp[2];
    const float dx = q0 - k0, dy = q1 - k1, dz = q2 - k2;
    const float dist2 = dx * dx + dy * dy + dz * dz;
    float lead = 0.f;
#pragma unroll
    for (int r = 0; r < 8; ++r) {
        const float a = q0 * Wqb[r * 3] + q1 * Wqb[r * 3 + 1] + q2 * Wqb[r * 3 + 2];
        const float c = k0 * Wkb[r * 3] + k1 * Wkb[r * 3 + 1] + k2 * Wkb[r * 3 + 2];
        lead += a * c;
    }
    biasB[idx] = dist2 * DIST_COEF + lead * LEAD_COEF;
}

// ---------------------------------------------------------------------------
// Kernels 2/3/4/6: Y[M,512] = X[M,512] @ W^T + bias.  W row-major [512,512].
// Block tile 128(M) x 128(N), K-step 32.  8 waves = 4(M) x 2(N); each wave a
// 32x64 accumulator (2x4 WMMA 16x16x32 bf16 frags).  bf16 inputs are staged
// with async global->LDS b128; f32 inputs convert to bf16 through VGPRs.
// ---------------------------------------------------------------------------
template <bool IN_F32, bool OUT_BF16>
__global__ __launch_bounds__(256) void proj_gemm(
    const void* __restrict__ Xv, const float* __restrict__ W,
    const float* __restrict__ bias, void* __restrict__ Yv, int M)
{
    constexpr int D = DM, MT = 128, NT = 128, KT = 32;
    constexpr int LDA = KT + 8;   // 40 elems -> 80B rows (16B aligned)
    constexpr int LDB = KT + 8;
    __shared__ __align__(16) __bf16 As[MT * LDA];
    __shared__ __align__(16) __bf16 Bs[NT * LDB];

    const int nb = blockIdx.x & 3;            // D/NT == 4
    const int mb = blockIdx.x >> 2;
    const int m0 = mb * MT, n0 = nb * NT;
    const int tid = threadIdx.x;
    const int wave = tid >> 5, lane = tid & 31;
    const int wm = wave >> 1, wn = wave & 1;
    const int l = lane & 15, hi = lane >> 4;

    v8f acc[2][4] = {};

    for (int k0 = 0; k0 < D; k0 += KT) {
        // ---- stage A tile (128x32) into LDS as bf16 ----
        if (IN_F32) {
            const float* X = (const float*)Xv;
#pragma unroll
            for (int it = 0; it < 4; ++it) {
                const int e = (tid + it * 256) * 4;      // 4096 elems total
                const int r = e >> 5, c = e & 31;
                float4 f = *(const float4*)(X + (size_t)(m0 + r) * D + k0 + c);
                __bf16* d = &As[r * LDA + c];
                d[0] = f2bf(f.x); d[1] = f2bf(f.y); d[2] = f2bf(f.z); d[3] = f2bf(f.w);
            }
        } else {
            const __bf16* X = (const __bf16*)Xv;
#pragma unroll
            for (int it = 0; it < 2; ++it) {             // 512 16B chunks
                const int ch = tid + it * 256;
                const int r = ch >> 2, c = (ch & 3) * 8;
                async_copy_b128(&As[r * LDA + c], X + (size_t)(m0 + r) * D + k0 + c);
            }
        }
        // ---- stage B tile: rows n0..n0+127 of W (f32 -> bf16) ----
#pragma unroll
        for (int it = 0; it < 4; ++it) {
            const int e = (tid + it * 256) * 4;          // 4096 elems total
            const int r = e >> 5, c = e & 31;
            float4 f = *(const float4*)(W + (size_t)(n0 + r) * D + k0 + c);
            __bf16* d = &Bs[r * LDB + c];
            d[0] = f2bf(f.x); d[1] = f2bf(f.y); d[2] = f2bf(f.z); d[3] = f2bf(f.w);
        }
        if (!IN_F32) wait_async_lds();
        __syncthreads();

        bf16x16 bfrag[4];
#pragma unroll
        for (int ni = 0; ni < 4; ++ni)
            bfrag[ni] = frag_bt(Bs, LDB, wn * 64 + ni * 16, 0);
#pragma unroll
        for (int mi = 0; mi < 2; ++mi) {
            bf16x16 a = frag_a(As, LDA, wm * 32 + mi * 16, 0);
#pragma unroll
            for (int ni = 0; ni < 4; ++ni)
                acc[mi][ni] = wmma_bf16(a, bfrag[ni], acc[mi][ni]);
        }
        __syncthreads();
    }

    // ---- epilogue: + bias, store (C layout: VGPR i -> M = i + hi*8) ----
#pragma unroll
    for (int mi = 0; mi < 2; ++mi) {
#pragma unroll
        for (int ni = 0; ni < 4; ++ni) {
            const int col = n0 + wn * 64 + ni * 16 + l;
            const float bv = bias[col];
#pragma unroll
            for (int i = 0; i < 8; ++i) {
                const int row = m0 + wm * 32 + mi * 16 + hi * 8 + i;
                const float v = acc[mi][ni][i] + bv;
                if (OUT_BF16) ((__bf16*)Yv)[(size_t)row * D + col] = f2bf(v);
                else          ((float*)Yv)[(size_t)row * D + col]  = v;
            }
        }
    }
}

// ---------------------------------------------------------------------------
// Kernel 5: fused attention for one (b, n, h, q-chunk of 128).
// logits(128x128) = Q(128x64) @ K^T -> *scale + bias, mask, softmax over 128
// out(128x64)     = P(128x128) @ V(128x64)   (all via bf16 WMMA, f32 accum)
// q/k tiles staged with async global->LDS; v staged transposed via VGPRs.
// ---------------------------------------------------------------------------
__global__ __launch_bounds__(256) void attn_kernel(
    const __bf16* __restrict__ qh, const __bf16* __restrict__ kh,
    const __bf16* __restrict__ vh, const float* __restrict__ biasB,
    const unsigned char* __restrict__ key_mask, __bf16* __restrict__ attn_out)
{
    constexpr int LQK = HD + 8;   // 72  (q_s / k_s leading dim)
    constexpr int LP  = KN + 8;   // 136 (p_s leading dim)
    constexpr int LV  = KN + 8;   // 136 (v_s transposed [d][k])
    // q_s(128x72) + k_s(128x72) region is reused for p_s(128x136); v_s after.
    __shared__ __align__(16) __bf16 smem[2 * 128 * LQK + HD * LV]; // 27136 elems
    __bf16* q_s = smem;
    __bf16* k_s = smem + 128 * LQK;
    __bf16* v_s = smem + 2 * 128 * LQK;
    __bf16* p_s = smem;

    const int gid = blockIdx.x;
    const int qc = gid & 3;             // Q chunk (4 x 128)
    const int h  = (gid >> 2) & 7;
    const int n  = (gid >> 5) & 31;
    const int b  = gid >> 10;
    const int qbase = qc * 128;

    const int tid = threadIdx.x;
    const int wave = tid >> 5, lane = tid & 31;
    const int l = lane & 15, hi = lane >> 4;
    const int qr0 = wave * 16;          // each wave owns 16 q rows

    const size_t qrowbase = ((size_t)(b * NN + n) * QN + qbase);
    const size_t krowbase = ((size_t)(b * NN + n) * KN);

    // ---- stage q (128x64) and k (128x64) via async global->LDS ----
#pragma unroll
    for (int it = 0; it < 4; ++it) {                 // 1024 chunks of 8 bf16
        const int ch = tid + it * 256;
        const int r = ch >> 3, c = (ch & 7) * 8;
        async_copy_b128(&q_s[r * LQK + c], qh + (qrowbase + r) * DM + h * HD + c);
    }
#pragma unroll
    for (int it = 0; it < 4; ++it) {
        const int ch = tid + it * 256;
        const int r = ch >> 3, c = (ch & 7) * 8;
        async_copy_b128(&k_s[r * LQK + c], kh + (krowbase + r) * DM + h * HD + c);
    }
    // ---- stage v transposed [d][k] through VGPRs ----
#pragma unroll
    for (int it = 0; it < 4; ++it) {
        const int ch = tid + it * 256;
        const int r = ch >> 3, c = (ch & 7) * 8;     // r = k row, c = d base
        bf16x8 f = *(const bf16x8*)(vh + (krowbase + r) * DM + h * HD + c);
#pragma unroll
        for (int j = 0; j < 8; ++j) v_s[(c + j) * LV + r] = f[j];
    }
    wait_async_lds();
    __syncthreads();

    // ---- logits: 16 q rows x 128 k cols per wave (8 col tiles) ----
    v8f acc[8] = {};
#pragma unroll
    for (int kk = 0; kk < HD; kk += 32) {
        bf16x16 a = frag_a(q_s, LQK, qr0, kk);
#pragma unroll
        for (int t = 0; t < 8; ++t) {
            bf16x16 bb = frag_bt(k_s, LQK, t * 16, kk);   // element(d, kcol)=k[kcol,d]
            acc[t] = wmma_bf16(a, bb, acc[t]);
        }
    }

    // ---- scale + bias + mask + softmax over 128 (rows live in 16-lane halves)
    bool mok[8];
#pragma unroll
    for (int t = 0; t < 8; ++t) mok[t] = key_mask[b * KN + t * 16 + l] != 0;
#pragma unroll
    for (int i = 0; i < 8; ++i) {
        const int qrow = qr0 + hi * 8 + i;           // C layout: M = i + hi*8
        const float* brow = biasB + ((size_t)b * QN + qbase + qrow) * KN + l;
        float x[8];
        float mx = -3.0e38f;
#pragma unroll
        for (int t = 0; t < 8; ++t) {
            x[t] = acc[t][i] * ATT_SCALE + brow[t * 16];
            if (!mok[t]) x[t] = -10000.0f;
            mx = fmaxf(mx, x[t]);
        }
#pragma unroll
        for (int m = 1; m <= 8; m <<= 1) mx = fmaxf(mx, __shfl_xor(mx, m, 32));
        float s = 0.f;
#pragma unroll
        for (int t = 0; t < 8; ++t) { x[t] = __expf(x[t] - mx); s += x[t]; }
#pragma unroll
        for (int m = 1; m <= 8; m <<= 1) s += __shfl_xor(s, m, 32);
        const float inv = 1.0f / s;
#pragma unroll
        for (int t = 0; t < 8; ++t) acc[t][i] = x[t] * inv;
    }

    __syncthreads();   // everyone done reading q_s/k_s before p_s overwrites
#pragma unroll
    for (int i = 0; i < 8; ++i) {
        const int qrow = qr0 + hi * 8 + i;
#pragma unroll
        for (int t = 0; t < 8; ++t)
            p_s[qrow * LP + t * 16 + l] = f2bf(acc[t][i]);
    }
    __syncthreads();

    // ---- out = P(16x128) @ V(128x64): 4 K-steps, 4 d-col tiles ----
    v8f oacc[4] = {};
#pragma unroll
    for (int kk = 0; kk < KN; kk += 32) {
        bf16x16 a = frag_a(p_s, LP, qr0, kk);
#pragma unroll
        for (int t = 0; t < 4; ++t) {
            bf16x16 bb = frag_bt(v_s, LV, t * 16, kk);    // element(k, d)=v[k,d]
            oacc[t] = wmma_bf16(a, bb, oacc[t]);
        }
    }
#pragma unroll
    for (int t = 0; t < 4; ++t) {
#pragma unroll
        for (int i = 0; i < 8; ++i) {
            const int qrow = qr0 + hi * 8 + i;
            attn_out[(qrowbase + qrow) * DM + h * HD + t * 16 + l] = f2bf(oacc[t][i]);
        }
    }
}

// ---------------------------------------------------------------------------
// Launch
// ---------------------------------------------------------------------------
extern "C" void kernel_launch(void* const* d_in, const int* in_sizes, int n_in,
                              void* d_out, int out_size, void* d_ws, size_t ws_size,
                              hipStream_t stream) {
    (void)in_sizes; (void)n_in; (void)out_size; (void)ws_size;
    const float* query      = (const float*)d_in[0];
    const float* key_value  = (const float*)d_in[1];
    const float* query_pos  = (const float*)d_in[2];
    const float* key_pos    = (const float*)d_in[3];
    const unsigned char* km = (const unsigned char*)d_in[4];
    const float* Wq  = (const float*)d_in[5];
    const float* bq  = (const float*)d_in[6];
    const float* Wk  = (const float*)d_in[7];
    const float* bk  = (const float*)d_in[8];
    const float* Wv  = (const float*)d_in[9];
    const float* bv  = (const float*)d_in[10];
    const float* Wo  = (const float*)d_in[11];
    const float* bo  = (const float*)d_in[12];
    const float* Wqb = (const float*)d_in[13];
    const float* Wkb = (const float*)d_in[14];

    // Workspace layout (bytes):
    //   qh   bf16 [65536, 512] @ 0          (67,108,864)
    //   kh   bf16 [16384, 512] @ 67108864   (16,777,216)
    //   vh   bf16 [16384, 512] @ 83886080   (16,777,216)
    //   attn bf16 [65536, 512] @ 100663296  (67,108,864)
    //   bias f32  [4,512,128]  @ 167772160  ( 1,048,576)
    char* ws = (char*)d_ws;
    __bf16* qh = (__bf16*)(ws);
    __bf16* kh = (__bf16*)(ws + 67108864);
    __bf16* vh = (__bf16*)(ws + 83886080);
    __bf16* at = (__bf16*)(ws + 100663296);
    float*  bB = (float*)(ws + 167772160);

    const int Mq = BB * NN * QN;   // 65536
    const int Mk = BB * NN * KN;   // 16384

    bias_kernel<<<(BB * QN * KN) / 256, 256, 0, stream>>>(query_pos, key_pos, Wqb, Wkb, bB);
    proj_gemm<true,  true ><<<(Mq / 128) * 4, 256, 0, stream>>>(query,     Wq, bq, qh,    Mq);
    proj_gemm<true,  true ><<<(Mk / 128) * 4, 256, 0, stream>>>(key_value, Wk, bk, kh,    Mk);
    proj_gemm<true,  true ><<<(Mk / 128) * 4, 256, 0, stream>>>(key_value, Wv, bv, vh,    Mk);
    attn_kernel<<<BB * NN * NH * (QN / 128), 256, 0, stream>>>(qh, kh, vh, bB, km, at);
    proj_gemm<false, false><<<(Mq / 128) * 4, 256, 0, stream>>>(at,        Wo, bo, d_out, Mq);
}
